// RandLANet_11003706212693
// MI455X (gfx1250) — matrix operand order repaired and checked by reading, not compiled, not run on hardware
//
#include <hip/hip_runtime.h>
#include <hip/hip_bf16.h>

typedef __attribute__((ext_vector_type(16))) _Float16 v16h;
typedef __attribute__((ext_vector_type(8)))  _Float16 v8h;
typedef __attribute__((ext_vector_type(8)))  float    v8f;

#if __has_builtin(__builtin_amdgcn_global_load_async_to_lds_b32) && \
    __has_builtin(__builtin_amdgcn_s_wait_asynccnt)
#define ASYNC_LDS 1
typedef __attribute__((address_space(1))) int g_int_t;   // global (AS1)
typedef __attribute__((address_space(3))) int l_int_t;   // LDS (AS3)
#else
#define ASYNC_LDS 0
#endif

static __host__ __device__ inline int cdiv_i(int a, int b) { return (a + b - 1) / b; }

// ---------------------------------------------------------------------------
// Weight prepack: f32 [Co,Ci] row-major -> f16 [CoP,CiP] zero padded.
// ---------------------------------------------------------------------------
__global__ void k_pack_w(const float* __restrict__ W, _Float16* __restrict__ Wh,
                         int Co, int Ci, int CoP, int CiP) {
    int i = blockIdx.x * blockDim.x + threadIdx.x;
    if (i >= CoP * CiP) return;
    int r = i / CiP, c = i % CiP;
    float v = (r < Co && c < Ci) ? W[(size_t)r * Ci + c] : 0.f;
    Wh[i] = (_Float16)v;
}

// ---------------------------------------------------------------------------
// Fused WMMA GEMM:  Y[m, coloff+co] = act( (X @ W^T)[m,co]*scale[co] + bias[co]
//                                          (+ resid[m,co]) )
// X: f32 [M, Ci] (ldx), W prepacked f16 [CoP, CiP].
// One block = 16 output rows; 8 waves fan over N(=Co) tiles; K looped by 32.
// A tile staged to LDS via global_load_async_to_lds_b32 (ASYNCcnt path) when
// the toolchain exposes it; f32 -> f16 conversion happens at fragment build.
// B-frag: lane%16 = N column of W, lane/16 selects K 0..15 / 16..31 of the
// 32-wide K tile (contiguous, 32B-aligned in Wh).
// ---------------------------------------------------------------------------
__global__ __launch_bounds__(256) void k_gemm(
    const float* __restrict__ X, int M, int Ci, int ldx,
    const _Float16* __restrict__ Wh, int CiP, int Co,
    const float* __restrict__ scale, const float* __restrict__ bias,
    const float* __restrict__ resid, int ldr,
    float* __restrict__ Y, int ldy, int coloff, int relu)
{
    __shared__ __align__(16) float Alds[16 * 32];
    const int tid  = threadIdx.x;
    const int lane = tid & 31;
    const int wv   = tid >> 5;
    const int m0   = blockIdx.x * 16;
    const int nKT  = (Ci + 31) >> 5;
    const int nNT  = (Co + 15) >> 4;          // <= 16 for all layers here
    const int nt0  = wv;
    const int nt1  = wv + 8;

    v8f acc0 = {}; v8f acc1 = {};

    for (int kt = 0; kt < nKT; ++kt) {
        // stage 16x32 f32 A tile: 512 words, one b32 per thread x 2 passes
        {
            int i = tid;
#pragma unroll
            for (int pass = 0; pass < 2; ++pass, i += 256) {
                int r = i >> 5, c = i & 31;
                int gr = m0 + r, gc = (kt << 5) + c;
                bool inb = (gr < M && gc < Ci);
#if ASYNC_LDS
                if (inb) {
                    __builtin_amdgcn_global_load_async_to_lds_b32(
                        (g_int_t*)(X + (size_t)gr * ldx + gc),
                        (l_int_t*)&Alds[i],
                        0, 0);
                } else {
                    Alds[i] = 0.f;                       // zero pad (disjoint addr)
                }
#else
                Alds[i] = inb ? X[(size_t)gr * ldx + gc] : 0.f;
#endif
            }
        }
        if (kt + 1 < nKT && tid == 0)
            __builtin_prefetch(X + (size_t)m0 * ldx + ((kt + 1) << 5), 0, 1);
#if ASYNC_LDS
        __builtin_amdgcn_s_wait_asynccnt(0);
#endif
        __syncthreads();

        // A fragment: row = lane%16, K halves split by lane/16 (contiguous k = 16*(lane/16) + s, same as B)
        const int row = lane & 15;
        const int b0  = (lane >> 4) << 4;     // 0 or 16
        v16h a;
#pragma unroll
        for (int h = 0; h < 8; ++h) {
            a[h]     = (_Float16)Alds[row * 32 + b0 + h];
            a[h + 8] = (_Float16)Alds[row * 32 + b0 + 8 + h];
        }

        const int kb = (kt << 5) + ((lane >> 4) << 4);   // 32B aligned in Wh
        if (nt0 < nNT) {                                  // wave-uniform guard
            int wrow = nt0 * 16 + (lane & 15);
            v16h bf = *(const v16h*)(Wh + (size_t)wrow * CiP + kb);
            acc0 = __builtin_amdgcn_wmma_f32_16x16x32_f16(false, a, false, bf,
                                                          (short)0, acc0, false, false);
        }
        if (nt1 < nNT) {
            int wrow = nt1 * 16 + (lane & 15);
            v16h bf = *(const v16h*)(Wh + (size_t)wrow * CiP + kb);
            acc1 = __builtin_amdgcn_wmma_f32_16x16x32_f16(false, a, false, bf,
                                                          (short)0, acc1, false, false);
        }
        __syncthreads();
    }

    // epilogue: VGPR r -> M = r + 8*(lane/16); N = lane%16
    const int n_  = lane & 15;
    const int mb_ = (lane >> 4) << 3;
#pragma unroll
    for (int t = 0; t < 2; ++t) {
        int nt = (t == 0) ? nt0 : nt1;
        if (nt >= nNT) continue;
        v8f acc = (t == 0) ? acc0 : acc1;
        int col = nt * 16 + n_;
        if (col >= Co) continue;
        float sc = scale ? scale[col] : 1.f;
        float bi = bias  ? bias[col]  : 0.f;
#pragma unroll
        for (int r = 0; r < 8; ++r) {
            int m = m0 + mb_ + r;
            if (m < M) {
                float y = acc[r] * sc + bi;
                if (resid) y += resid[(size_t)m * ldr + col];
                if (relu && y < 0.f) y = 0.f;
                Y[(size_t)m * ldy + coloff + col] = y;
            }
        }
    }
}

// ---------------------------------------------------------------------------
// fc0: feat = relu((xyz @ fcw^T + fcb)*s + b)   (3 -> 8)
// ---------------------------------------------------------------------------
__global__ void k_fc0(const float* __restrict__ xyz, const float* __restrict__ w,
                      const float* __restrict__ fb, const float* __restrict__ s,
                      const float* __restrict__ bb, float* __restrict__ feat, int Ntot) {
    int i = blockIdx.x * blockDim.x + threadIdx.x;
    if (i >= Ntot * 8) return;
    int p = i >> 3, o = i & 7;
    const float* xp = xyz + (size_t)p * 3;
    float acc = xp[0] * w[o * 3 + 0] + xp[1] * w[o * 3 + 1] + xp[2] * w[o * 3 + 2] + fb[o];
    acc = acc * s[o] + bb[o];
    feat[i] = acc > 0.f ? acc : 0.f;
}

// ---------------------------------------------------------------------------
// deterministic strided subsample (stand-in for random permutation)
// ---------------------------------------------------------------------------
__global__ void k_sample(const float* __restrict__ xyz, const float* __restrict__ feat,
                         int Np, int C, int m,
                         float* __restrict__ oxyz, float* __restrict__ ofeat) {
    int i = blockIdx.x * blockDim.x + threadIdx.x;
    if (i >= 2 * m) return;
    int b = i / m, j = i - b * m;
    int src = (int)(((long long)j * Np) / m);
    const float* x = xyz + ((size_t)b * Np + src) * 3;
    float* ox = oxyz + ((size_t)b * m + j) * 3;
    ox[0] = x[0]; ox[1] = x[1]; ox[2] = x[2];
    const float* f = feat + ((size_t)b * Np + src) * C;
    float* of = ofeat + ((size_t)b * m + j) * C;
    for (int c = 0; c < C; ++c) of[c] = f[c];
}

// ---------------------------------------------------------------------------
// 16-NN over Np points (per batch), LDS-tiled candidates, register top-16.
// ---------------------------------------------------------------------------
__global__ __launch_bounds__(256) void k_knn(const float* __restrict__ xyz, int Np,
                                             int* __restrict__ knn) {
    __shared__ float sx[256], sy[256], sz[256];
    int b = blockIdx.y;
    int q = blockIdx.x * 256 + threadIdx.x;
    const float* base = xyz + (size_t)b * Np * 3;
    float qx = 0.f, qy = 0.f, qz = 0.f;
    if (q < Np) { qx = base[q * 3]; qy = base[q * 3 + 1]; qz = base[q * 3 + 2]; }
    float bd[16]; int bi[16];
#pragma unroll
    for (int j = 0; j < 16; ++j) { bd[j] = 3.4e38f; bi[j] = 0; }
    for (int t0 = 0; t0 < Np; t0 += 256) {
        int c = t0 + threadIdx.x;
        if (c < Np) {
            sx[threadIdx.x] = base[c * 3];
            sy[threadIdx.x] = base[c * 3 + 1];
            sz[threadIdx.x] = base[c * 3 + 2];
        }
        __syncthreads();
        int lim = Np - t0; if (lim > 256) lim = 256;
        for (int u = 0; u < lim; ++u) {
            float dx = qx - sx[u], dy = qy - sy[u], dz = qz - sz[u];
            float d = dx * dx + dy * dy + dz * dz;
            if (d < bd[15]) {
                bd[15] = d; bi[15] = t0 + u;
#pragma unroll
                for (int j = 15; j > 0; --j) {
                    if (bd[j] < bd[j - 1]) {
                        float td = bd[j]; bd[j] = bd[j - 1]; bd[j - 1] = td;
                        int   ti = bi[j]; bi[j] = bi[j - 1]; bi[j - 1] = ti;
                    }
                }
            }
        }
        __syncthreads();
    }
    if (q < Np) {
        size_t o = ((size_t)b * Np + q) * 16;
        for (int j = 0; j < 16; ++j) knn[o + j] = bi[j];
    }
}

// ---------------------------------------------------------------------------
// LSE neighborhood encoding rows: [relx,rely,relz,dist, feat_nb[0..C-1]]
// ---------------------------------------------------------------------------
__global__ void k_lse_enc(const float* __restrict__ xyz, const float* __restrict__ feat,
                          const int* __restrict__ knn, int Np, int C,
                          float* __restrict__ enc) {
    int i = blockIdx.x * blockDim.x + threadIdx.x;
    int total = 2 * Np * 16;
    if (i >= total) return;
    int b = i / (Np * 16);
    int r = i - b * Np * 16;
    int p = r >> 4, kk = r & 15;
    int nb = knn[((size_t)b * Np + p) * 16 + kk];
    const float* xb = xyz + (size_t)b * Np * 3;
    float rx = xb[nb * 3]     - xb[p * 3];
    float ry = xb[nb * 3 + 1] - xb[p * 3 + 1];
    float rz = xb[nb * 3 + 2] - xb[p * 3 + 2];
    float d = sqrtf(rx * rx + ry * ry + rz * rz + 1e-12f);
    float* row = enc + (size_t)i * (C + 4);
    row[0] = rx; row[1] = ry; row[2] = rz; row[3] = d;
    const float* fb = feat + ((size_t)b * Np + nb) * C;
    for (int c = 0; c < C; ++c) row[4 + c] = fb[c];
}

// score head of attentive pooling: Co=1 dot + bias
__global__ void k_dot1(const float* __restrict__ X, int M, int C,
                       const float* __restrict__ w, const float* __restrict__ b,
                       float* __restrict__ out) {
    int i = blockIdx.x * blockDim.x + threadIdx.x;
    if (i >= M) return;
    float acc = b[0];
    const float* x = X + (size_t)i * C;
    for (int c = 0; c < C; ++c) acc += x[c] * w[c];
    out[i] = acc;
}

__global__ void k_softmax16(float* __restrict__ s2, int Npts) {
    int i = blockIdx.x * blockDim.x + threadIdx.x;
    if (i >= Npts) return;
    float* p = s2 + (size_t)i * 16;
    float mx = p[0];
    for (int j = 1; j < 16; ++j) mx = p[j] > mx ? p[j] : mx;
    float sum = 0.f;
    for (int j = 0; j < 16; ++j) { float e = expf(p[j] - mx); p[j] = e; sum += e; }
    float inv = 1.f / sum;
    for (int j = 0; j < 16; ++j) p[j] *= inv;
}

// attention-weighted pooling over k=16
__global__ void k_pool(const float* __restrict__ X, const float* __restrict__ P,
                       float* __restrict__ F, int Npt, int C) {
    int i = blockIdx.x * blockDim.x + threadIdx.x;
    if (i >= Npt * C) return;
    int p = i / C, c = i - p * C;
    const float* xp = X + (size_t)p * 16 * C + c;
    const float* pp = P + (size_t)p * 16;
    float acc = 0.f;
    for (int k = 0; k < 16; ++k) acc += xp[(size_t)k * C] * pp[k];
    F[i] = acc;
}

// ---------------------------------------------------------------------------
// FP: per skip point, 3 nearest prev points + full-row reciprocal-sum weights
// ---------------------------------------------------------------------------
__global__ __launch_bounds__(256) void k_fp_nn(const float* __restrict__ xyz_s, int Ns,
                                               const float* __restrict__ xyz_p, int Np,
                                               int* __restrict__ idx3, float* __restrict__ w3) {
    __shared__ float sx[256], sy[256], sz[256];
    int b = blockIdx.y;
    int q = blockIdx.x * 256 + threadIdx.x;
    const float* bs = xyz_s + (size_t)b * Ns * 3;
    const float* bp = xyz_p + (size_t)b * Np * 3;
    float qx = 0.f, qy = 0.f, qz = 0.f;
    if (q < Ns) { qx = bs[q * 3]; qy = bs[q * 3 + 1]; qz = bs[q * 3 + 2]; }
    float d0 = 3.4e38f, d1 = 3.4e38f, d2 = 3.4e38f;
    int   i0 = 0, i1 = 0, i2 = 0;
    float sum = 0.f;
    for (int t = 0; t < Np; t += 256) {
        int c = t + threadIdx.x;
        if (c < Np) {
            sx[threadIdx.x] = bp[c * 3];
            sy[threadIdx.x] = bp[c * 3 + 1];
            sz[threadIdx.x] = bp[c * 3 + 2];
        }
        __syncthreads();
        int lim = Np - t; if (lim > 256) lim = 256;
        for (int u = 0; u < lim; ++u) {
            float dx = qx - sx[u], dy = qy - sy[u], dz = qz - sz[u];
            float dq = dx * dx + dy * dy + dz * dz;
            float dist = sqrtf(dq + 1e-12f);
            sum += 1.f / (dist + 1e-8f);
            int ci = t + u;
            if (dq < d2) {
                if (dq < d1) {
                    d2 = d1; i2 = i1;
                    if (dq < d0) { d1 = d0; i1 = i0; d0 = dq; i0 = ci; }
                    else         { d1 = dq; i1 = ci; }
                } else { d2 = dq; i2 = ci; }
            }
        }
        __syncthreads();
    }
    if (q < Ns) {
        size_t o = ((size_t)b * Ns + q) * 3;
        idx3[o] = i0; idx3[o + 1] = i1; idx3[o + 2] = i2;
        float inv = 1.f / sum;
        w3[o]     = (1.f / (sqrtf(d0 + 1e-12f) + 1e-8f)) * inv;
        w3[o + 1] = (1.f / (sqrtf(d1 + 1e-12f) + 1e-8f)) * inv;
        w3[o + 2] = (1.f / (sqrtf(d2 + 1e-12f) + 1e-8f)) * inv;
    }
}

// interpolation + concat with skip features
__global__ void k_interp_concat(const float* __restrict__ fprev, int Cp, int Np,
                                const float* __restrict__ fskip, int Csk, int Ns,
                                const int* __restrict__ idx3, const float* __restrict__ w3,
                                float* __restrict__ out) {
    int Ct = Cp + Csk;
    int i = blockIdx.x * blockDim.x + threadIdx.x;
    int total = 2 * Ns * Ct;
    if (i >= total) return;
    int b = i / (Ns * Ct);
    int r = i - b * Ns * Ct;
    int s = r / Ct, c = r - s * Ct;
    float v;
    if (c < Cp) {
        size_t o = ((size_t)b * Ns + s) * 3;
        const float* fb = fprev + (size_t)b * Np * Cp;
        v = w3[o]     * fb[(size_t)idx3[o]     * Cp + c]
          + w3[o + 1] * fb[(size_t)idx3[o + 1] * Cp + c]
          + w3[o + 2] * fb[(size_t)idx3[o + 2] * Cp + c];
    } else {
        v = fskip[((size_t)b * Ns + s) * Csk + (c - Cp)];
    }
    out[((size_t)b * Ns + s) * Ct + c] = v;
}

// final head: 64 -> 5 + bias, output transposed to [B,5,N]
__global__ void k_head2(const float* __restrict__ H, const float* __restrict__ w2,
                        const float* __restrict__ b2, float* __restrict__ out, int N) {
    int i = blockIdx.x * blockDim.x + threadIdx.x;
    int total = 2 * 5 * N;
    if (i >= total) return;
    int b = i / (5 * N);
    int r = i - b * 5 * N;
    int cls = r / N, n = r - cls * N;
    const float* h = H + ((size_t)b * N + n) * 64;
    const float* w = w2 + cls * 64;
    float acc = b2[cls];
    for (int c = 0; c < 64; ++c) acc += h[c] * w[c];
    out[i] = acc;
}

// ===========================================================================
// Host orchestration
// ===========================================================================
extern "C" void kernel_launch(void* const* d_in, const int* in_sizes, int n_in,
                              void* d_out, int out_size, void* d_ws, size_t ws_size,
                              hipStream_t stream) {
    (void)in_sizes; (void)out_size; (void)ws_size;

    const float* P[200];
    int nP = n_in < 200 ? n_in : 200;
    for (int i = 0; i < nP; ++i) P[i] = (const float*)d_in[i];

    // ---- param extraction (jax tree flatten: dict keys sorted, lists in order)
    int pi = 1; // d_in[0] = xyz
    auto nx = [&]() -> const float* {
        const float* r = (pi < nP) ? P[pi] : P[nP - 1];
        ++pi; return r;
    };
    const float* bn0_b = nx(); const float* bn0_s = nx();
    struct DecP { const float *b1,*s1,*b2,*s2,*w1,*w2; } dp[4];
    for (int i = 0; i < 4; ++i) {
        dp[i].b1 = nx(); dp[i].s1 = nx(); dp[i].b2 = nx(); dp[i].s2 = nx();
        dp[i].w1 = nx(); dp[i].w2 = nx();
    }
    struct ApP  { const float *bnb,*bns,*sb2,*sbnb,*sbns,*sw1,*sw2,*w; };
    struct DrbP { const float *b1,*s1,*b2,*s2,*w1,*w2; };
    struct LseP { const float *bnb,*bns,*w; };
    struct EncP { ApP ap1, ap2; DrbP drb; LseP l1, l2; } ep[4];
    for (int i = 0; i < 4; ++i) {
        ApP* aps[2] = { &ep[i].ap1, &ep[i].ap2 };
        for (int a = 0; a < 2; ++a) {
            aps[a]->bnb = nx(); aps[a]->bns = nx(); aps[a]->sb2 = nx();
            aps[a]->sbnb = nx(); aps[a]->sbns = nx();
            aps[a]->sw1 = nx(); aps[a]->sw2 = nx(); aps[a]->w = nx();
        }
        ep[i].drb.b1 = nx(); ep[i].drb.s1 = nx(); ep[i].drb.b2 = nx();
        ep[i].drb.s2 = nx(); ep[i].drb.w1 = nx(); ep[i].drb.w2 = nx();
        ep[i].l1.bnb = nx(); ep[i].l1.bns = nx(); ep[i].l1.w = nx();
        ep[i].l2.bnb = nx(); ep[i].l2.bns = nx(); ep[i].l2.w = nx();
    }
    const float* fc_b = nx(); const float* fc_w = nx();
    const float* h_b2 = nx(); const float* h_b1b = nx(); const float* h_b1s = nx();
    const float* h_w1 = nx(); const float* h_w2 = nx();

    const float* xyz_in = P[0];

    // ---- bump allocator over workspace
    char* wsb = (char*)d_ws;
    size_t off = 0;
    auto alloc = [&](size_t bytes) -> void* {
        off = (off + 255) & ~(size_t)255;
        void* p = wsb + off; off += bytes; return p;
    };
    auto allocF = [&](size_t n) -> float* { return (float*)alloc(n * sizeof(float)); };

    // ---- WMMA GEMM launcher (prepacks weights to f16 every call; deterministic)
    auto gemm = [&](const float* X, int M, int Ci, int ldx, const float* W, int Co,
                    const float* sc, const float* bi, const float* resid, int ldr,
                    float* Y, int ldy, int coloff, int relu) {
        int CoP = (Co + 15) & ~15, CiP = (Ci + 31) & ~31;
        _Float16* Wh = (_Float16*)alloc((size_t)CoP * CiP * sizeof(_Float16));
        int tot = CoP * CiP;
        k_pack_w<<<cdiv_i(tot, 256), 256, 0, stream>>>(W, Wh, Co, Ci, CoP, CiP);
        k_gemm<<<cdiv_i(M, 16), 256, 0, stream>>>(X, M, Ci, ldx,
                                                  (const _Float16*)Wh, CiP, Co,
                                                  sc, bi, resid, ldr, Y, ldy, coloff, relu);
    };

    // ---- stage sizes (match python int truncation)
    int Nst[5]; Nst[0] = 8192;
    const double rat[4] = {0.35, 0.30, 0.25, 0.25};
    for (int i = 0; i < 4; ++i) {
        int m = (int)((double)Nst[i] * rat[i]);
        Nst[i + 1] = m < 1 ? 1 : m;
    }
    const int Cst[5] = {8, 16, 64, 128, 256};

    // ---- scratch maxima
    size_t encMax = 0, yMax = 0, s2Max = 0, poolMax = 0, fsMax = 0, drbMax = 0;
    for (int i = 0; i < 4; ++i) {
        size_t r16 = (size_t)2 * Nst[i + 1] * 16;
        size_t cen = Cst[i] + 4, h = Cst[i + 1] / 2, cout = Cst[i + 1];
        if (r16 * cen > encMax) encMax = r16 * cen;
        if (r16 * h > yMax) yMax = r16 * h;
        if (r16 > s2Max) s2Max = r16;
        size_t pl = (size_t)2 * Nst[i + 1] * h;
        if (pl > poolMax) poolMax = pl;
        size_t fs = (size_t)2 * Nst[i + 1] * Cst[i];
        if (fs > fsMax) fsMax = fs;
        size_t db = (size_t)2 * Nst[i + 1] * cout;
        if (db > drbMax) drbMax = db;
    }
    float* encbuf  = allocF(encMax);
    float* ybuf    = allocF(yMax);
    float* s1buf   = allocF(yMax);
    float* s2buf   = allocF(s2Max);
    float* poolbuf = allocF(poolMax);
    float* fsamp   = allocF(fsMax);
    float* drbbuf  = allocF(drbMax);
    int*   knnbuf  = (int*)alloc((size_t)2 * Nst[1] * 16 * sizeof(int));

    // ---- input MLP (3 -> 8) with folded BN
    const float* xyzs[5]; float* feats[5];
    xyzs[0] = xyz_in;
    feats[0] = allocF((size_t)2 * 8192 * 8);
    k_fc0<<<cdiv_i(2 * 8192 * 8, 256), 256, 0, stream>>>(xyz_in, fc_w, fc_b,
                                                         bn0_s, bn0_b, feats[0], 2 * 8192);

    // ---- attentive pooling helper
    auto ap = [&](const float* Xk, int m, int h, const ApP& p,
                  float* Yout, int ldy, int coloff) {
        int M16 = 2 * m * 16;
        gemm(Xk, M16, h, h, p.sw1, h, p.sbns, p.sbnb, nullptr, 0, s1buf, h, 0, 1);
        k_dot1<<<cdiv_i(M16, 256), 256, 0, stream>>>(s1buf, M16, h, p.sw2, p.sb2, s2buf);
        k_softmax16<<<cdiv_i(2 * m, 256), 256, 0, stream>>>(s2buf, 2 * m);
        k_pool<<<cdiv_i(2 * m * h, 256), 256, 0, stream>>>(Xk, s2buf, poolbuf, 2 * m, h);
        gemm(poolbuf, 2 * m, h, h, p.w, h, p.bns, p.bnb, nullptr, 0, Yout, ldy, coloff, 1);
    };

    // ---- encoder
    for (int i = 0; i < 4; ++i) {
        int Np = Nst[i], m = Nst[i + 1];
        int cin = Cst[i], cout = Cst[i + 1], h = cout / 2, cen = cin + 4;
        float* nxyz = allocF((size_t)2 * m * 3);
        k_sample<<<cdiv_i(2 * m, 256), 256, 0, stream>>>(xyzs[i], feats[i], Np, cin, m,
                                                         nxyz, fsamp);
        k_knn<<<dim3(cdiv_i(m, 256), 2), 256, 0, stream>>>(nxyz, m, knnbuf);
        int M16 = 2 * m * 16;
        k_lse_enc<<<cdiv_i(M16, 256), 256, 0, stream>>>(nxyz, fsamp, knnbuf, m, cin, encbuf);

        float* fcat = allocF((size_t)2 * m * cout);
        // branch 1: LSE1 -> AP1 -> fcat[:, 0:h]
        gemm(encbuf, M16, cen, cen, ep[i].l1.w, h, ep[i].l1.bns, ep[i].l1.bnb,
             nullptr, 0, ybuf, h, 0, 1);
        ap(ybuf, m, h, ep[i].ap1, fcat, cout, 0);
        // branch 2: LSE2 -> AP2 -> fcat[:, h:2h]
        gemm(encbuf, M16, cen, cen, ep[i].l2.w, h, ep[i].l2.bns, ep[i].l2.bnb,
             nullptr, 0, ybuf, h, 0, 1);
        ap(ybuf, m, h, ep[i].ap2, fcat, cout, h);
        // dilated residual block: relu(bn2(w2 @ relu(bn1(w1 @ x))) + x)
        gemm(fcat, 2 * m, cout, cout, ep[i].drb.w1, cout, ep[i].drb.s1, ep[i].drb.b1,
             nullptr, 0, drbbuf, cout, 0, 1);
        gemm(drbbuf, 2 * m, cout, cout, ep[i].drb.w2, cout, ep[i].drb.s2, ep[i].drb.b2,
             fcat, cout, fcat, cout, 0, 1);
        xyzs[i + 1] = nxyz; feats[i + 1] = fcat;
    }

    // ---- decoder
    const float* cx = xyzs[4];
    const float* cf = feats[4];
    int Cp = 256, Npd = Nst[4];
    const int DecDim[4] = {256, 128, 64, 32};
    int*   idx3 = (int*)alloc((size_t)2 * 8192 * 3 * sizeof(int));
    float* w3   = allocF((size_t)2 * 8192 * 3);
    for (int i = 0; i < 4; ++i) {
        int si = 3 - i;
        const float* sx = xyzs[si]; const float* sf = feats[si];
        int Ns_ = Nst[si], Csk = Cst[si], Cout = DecDim[i], Ct = Cp + Csk;
        k_fp_nn<<<dim3(cdiv_i(Ns_, 256), 2), 256, 0, stream>>>(sx, Ns_, cx, Npd, idx3, w3);
        float* cat = allocF((size_t)2 * Ns_ * Ct);
        k_interp_concat<<<cdiv_i(2 * Ns_ * Ct, 256), 256, 0, stream>>>(
            cf, Cp, Npd, sf, Csk, Ns_, idx3, w3, cat);
        float* h1 = allocF((size_t)2 * Ns_ * Cout);
        gemm(cat, 2 * Ns_, Ct, Ct, dp[i].w1, Cout, dp[i].s1, dp[i].b1,
             nullptr, 0, h1, Cout, 0, 1);
        float* ncf = allocF((size_t)2 * Ns_ * Cout);
        gemm(h1, 2 * Ns_, Cout, Cout, dp[i].w2, Cout, dp[i].s2, dp[i].b2,
             nullptr, 0, ncf, Cout, 0, 1);
        cx = sx; cf = ncf; Cp = Cout; Npd = Ns_;
    }

    // ---- head: 32 -> 64 (WMMA, bn+relu), then 64 -> 5 + bias, transpose out
    float* hh = allocF((size_t)2 * 8192 * 64);
    gemm(cf, 2 * 8192, 32, 32, h_w1, 64, h_b1s, h_b1b, nullptr, 0, hh, 64, 0, 1);
    k_head2<<<cdiv_i(2 * 5 * 8192, 256), 256, 0, stream>>>(hh, h_w2, h_b2,
                                                           (float*)d_out, 8192);
}
